// dmcnn_t_54090818125831
// MI455X (gfx1250) — matrix operand order repaired and checked by reading, not compiled, not run on hardware
//
#include <hip/hip_runtime.h>
#include <hip/hip_bf16.h>
#include <math.h>

// ---------------------------------------------------------------------------
// Types for CDNA5 WMMA
// ---------------------------------------------------------------------------
typedef __attribute__((ext_vector_type(16))) __bf16 v16bf;
typedef __attribute__((ext_vector_type(8)))  __bf16 v8bf;
typedef __attribute__((ext_vector_type(8)))  float  v8f;
typedef __attribute__((ext_vector_type(8)))  unsigned short v8us;

#define BDIM 64
#define SDIM 512
#define HDIM 768
#define FDIM 200
#define FPAD 208      // 13 WMMA N-tiles of 16
#define KDIM 2304     // 3 * 768
#define TDIM 34
#define NTILES 13

__device__ __forceinline__ unsigned short f2bf(float x) {
    unsigned u = __float_as_uint(x);
    u += 0x7FFFu + ((u >> 16) & 1u);          // round-to-nearest-even
    return (unsigned short)(u >> 16);
}

// ---------------------------------------------------------------------------
// Kernel 0: bulk convert emb fp32 -> bf16 (one pass; bf16 copy is L2-resident)
// ---------------------------------------------------------------------------
__global__ void convert_emb_kernel(const float* __restrict__ emb,
                                   unsigned short* __restrict__ embB) {
    const int v = blockIdx.x * blockDim.x + threadIdx.x;   // 8 elements each
    const int nvec = (BDIM * SDIM * HDIM) / 8;
    if (v >= nvec) return;
    const v8f x = *(const v8f*)(emb + (size_t)v * 8);
    v8us y;
#pragma unroll
    for (int i = 0; i < 8; ++i) y[i] = f2bf(x[i]);
    *(v8us*)(embB + (size_t)v * 8) = y;
}

// ---------------------------------------------------------------------------
// Kernel 1: build transposed bf16 weight matrix Bt[FPAD][KDIM],
//           Bt[n][w*768+h] = conv_w[n][h][w]   (conv_w is [F][H][W])
// ---------------------------------------------------------------------------
__global__ void build_bt_kernel(const float* __restrict__ conv_w,
                                unsigned short* __restrict__ Bt) {
    int idx = blockIdx.x * blockDim.x + threadIdx.x;
    if (idx >= FPAD * KDIM) return;
    int n = idx / KDIM;
    int k = idx - n * KDIM;
    int w = k / HDIM;
    int h = k - w * HDIM;
    float v = (n < FDIM) ? conv_w[(n * HDIM + h) * 3 + w] : 0.0f;
    Bt[idx] = f2bf(v);
}

// ---------------------------------------------------------------------------
// Kernel 2: conv-as-GEMM with v_wmma_f32_16x16x32_bf16.
// One wave (32 threads) per 16-row M-tile; 13 N-tile accumulators in regs.
// PRE=true : A loaded directly as bf16 from embB (no per-kstep conversion).
// PRE=false: A loaded fp32 + converted on the fly (workspace-lean fallback).
// ---------------------------------------------------------------------------
template <bool PRE>
__global__ __launch_bounds__(32)
void conv_wmma_kernel(const float* __restrict__ emb,
                      const unsigned short* __restrict__ embB,
                      const unsigned short* __restrict__ Bt,
                      const float* __restrict__ conv_b,
                      float* __restrict__ convOut) {
    const int lane  = threadIdx.x;                 // wave32
    const int mt    = blockIdx.x;                  // 0..2047
    const int mrowA = mt * 16 + (lane & 15);       // A row handled by this lane
    const int b     = mrowA >> 9;                  // /512
    const int sA    = mrowA & 511;
    const int kbA   = (lane < 16) ? 0 : 16;        // A layout: K {0..15} | {16..31}
    const int kbB   = (lane < 16) ? 0 : 16;        // B layout: K {0..15} | {16..31}
    const int nIdx  = lane & 15;

    v8f acc[NTILES] = {};

    for (int w = 0; w < 3; ++w) {
        const int srow   = sA + w - 1;
        const bool valid = (srow >= 0) && (srow < SDIM);
        const int  srowC = valid ? srow : 0;
        const float*          arowF = emb  + ((size_t)b * SDIM + srowC) * HDIM;
        const unsigned short* arowH = embB + ((size_t)b * SDIM + srowC) * HDIM;

        for (int h0 = 0; h0 < HDIM; h0 += 32) {
            const int k = w * HDIM + h0;

            // ---- A fragment (16x32 bf16), halo rows are zero ----
            union { v16bf v; v8bf h[2]; unsigned long long z[4]; } ta;
            if (valid) {
                if (PRE) {
                    ta.h[0] = *(const v8bf*)(arowH + h0 + kbA);        // K kb..kb+7
                    ta.h[1] = *(const v8bf*)(arowH + h0 + kbA + 8);    // K kb+8..kb+15
                } else {
                    const v8f lo = *(const v8f*)(arowF + h0 + kbA);
                    const v8f hi = *(const v8f*)(arowF + h0 + kbA + 8);
                    union { v8bf v; unsigned short u[8]; } c0, c1;
#pragma unroll
                    for (int i = 0; i < 8; ++i) { c0.u[i] = f2bf(lo[i]); c1.u[i] = f2bf(hi[i]); }
                    ta.h[0] = c0.v; ta.h[1] = c1.v;
                }
            } else {
#pragma unroll
                for (int i = 0; i < 4; ++i) ta.z[i] = 0ull;
            }
            const v16bf av = ta.v;

            // ---- 13 B fragments against the same A fragment ----
#pragma unroll
            for (int n = 0; n < NTILES; ++n) {
                const unsigned short* bp =
                    Bt + (size_t)(n * 16 + nIdx) * KDIM + k + kbB;
                const v16bf bv = *(const v16bf*)bp;
                acc[n] = __builtin_amdgcn_wmma_f32_16x16x32_bf16(
                    false, av, false, bv, (short)0, acc[n], false, false);
            }
        }
    }

    // ---- epilogue: bias + tanh, scatter per C-layout ----
#pragma unroll
    for (int n = 0; n < NTILES; ++n) {
        const int f = n * 16 + nIdx;
        if (f < FDIM) {
            const float bias = conv_b[f];
#pragma unroll
            for (int r = 0; r < 8; ++r) {
                const int m  = mt * 16 + r + ((lane < 16) ? 0 : 8);
                const int bo = m >> 9;
                const int so = m & 511;
                convOut[((size_t)bo * SDIM + so) * FDIM + f] = tanhf(acc[n][r] + bias);
            }
        }
    }
}

// ---------------------------------------------------------------------------
// Kernel 3: suffix cummax into rcm, then prefix cummax in-place (conv -> cm).
// 8-wide s-batches keep 8 loads in flight per wave (scan chain is only on the
// running max, not on the loads) -> bandwidth-bound instead of latency-bound.
// ---------------------------------------------------------------------------
__global__ void cummax_kernel(float* __restrict__ conv,   // in: conv, out: cm
                              float* __restrict__ rcm) {
    const int bb = blockIdx.x;
    const int f  = threadIdx.x;
    if (f >= FDIM) return;
    const size_t base = (size_t)bb * SDIM * FDIM + f;

    float run = -INFINITY;
    for (int s0 = SDIM - 8; s0 >= 0; s0 -= 8) {           // backward, batches of 8
        float v[8];
#pragma unroll
        for (int j = 0; j < 8; ++j) v[j] = conv[base + (size_t)(s0 + j) * FDIM];
        float r[8];
#pragma unroll
        for (int j = 7; j >= 0; --j) { run = fmaxf(run, v[j]); r[j] = run; }
#pragma unroll
        for (int j = 0; j < 8; ++j) rcm[base + (size_t)(s0 + j) * FDIM] = r[j];
    }

    run = -INFINITY;
    for (int s0 = 0; s0 < SDIM; s0 += 8) {                // forward, batches of 8
        float v[8];
#pragma unroll
        for (int j = 0; j < 8; ++j) v[j] = conv[base + (size_t)(s0 + j) * FDIM];
        float r[8];
#pragma unroll
        for (int j = 0; j < 8; ++j) { run = fmaxf(run, v[j]); r[j] = run; }
#pragma unroll
        for (int j = 0; j < 8; ++j) conv[base + (size_t)(s0 + j) * FDIM] = r[j];  // cm
    }
}

// ---------------------------------------------------------------------------
// Kernel 4: batch-mixing gather via GLOBAL_LOAD_ASYNC_TO_LDS + 400->34 linear.
// One block per (b, i) output row.
// ---------------------------------------------------------------------------
__global__ __launch_bounds__(128)
void logits_kernel(const float* __restrict__ cm,
                   const float* __restrict__ rcm,
                   const float* __restrict__ L_w,   // [34][400]
                   const float* __restrict__ L_b,   // [34]
                   float* __restrict__ out) {       // [B][S][34]
    __shared__ float sp[2 * FDIM];
    const int row = blockIdx.x;          // b*512 + i
    const int i   = row & 511;
    const int b   = row >> 9;
    const int tid = threadIdx.x;

    // Per-lane gathered async copy: global -> LDS, tracked by ASYNCcnt.
    for (int k = tid; k < 2 * FDIM; k += 128) {
        const float* gptr;
        if (i < SDIM - 1) {
            const int flat = b * (2 * FDIM) + k;     // torch cat.t().view mixing
            const int f    = flat >> 7;              // / 128
            const int b2   = flat & 127;             // % 128
            gptr = (b2 < BDIM)
                 ? (cm  + ((size_t)b2 * SDIM + i) * FDIM + f)
                 : (rcm + ((size_t)(b2 - BDIM) * SDIM + (i + 1)) * FDIM + f);
        } else {
            const int f = (k < FDIM) ? k : (k - FDIM);
            gptr = cm + ((size_t)b * SDIM + (SDIM - 1)) * FDIM + f;
        }
        const unsigned lds_off = (unsigned)(uintptr_t)(&sp[k]);  // addr[31:0] == LDS offset
        const unsigned long long ga = (unsigned long long)(uintptr_t)gptr;
        asm volatile("global_load_async_to_lds_b32 %0, %1, off"
                     :: "v"(lds_off), "v"(ga) : "memory");
    }
    asm volatile("s_wait_asynccnt 0x0" ::: "memory");
    __syncthreads();

    if (tid < TDIM) {
        float acc = L_b[tid];
        const float* wrow = L_w + (size_t)tid * (2 * FDIM);
#pragma unroll 4
        for (int k = 0; k < 2 * FDIM; ++k) acc = fmaf(sp[k], wrow[k], acc);
        out[(size_t)row * TDIM + tid] = acc;
    }
}

// ---------------------------------------------------------------------------
// Launch
// ---------------------------------------------------------------------------
extern "C" void kernel_launch(void* const* d_in, const int* in_sizes, int n_in,
                              void* d_out, int out_size, void* d_ws, size_t ws_size,
                              hipStream_t stream) {
    const float* emb    = (const float*)d_in[0];   // [64,512,768]
    const float* conv_w = (const float*)d_in[1];   // [200,768,3]
    const float* conv_b = (const float*)d_in[2];   // [200]
    const float* L_w    = (const float*)d_in[3];   // [34,400]
    const float* L_b    = (const float*)d_in[4];   // [34]
    float* out = (float*)d_out;                    // [64,512,34]

    char* ws = (char*)d_ws;
    const size_t convBytes = (size_t)BDIM * SDIM * FDIM * sizeof(float);      // 26,214,400
    const size_t btBytes   = (size_t)FPAD * KDIM * sizeof(unsigned short);    //    958,464
    const size_t embBytes  = (size_t)BDIM * SDIM * HDIM * sizeof(unsigned short); // 50,331,648

    float*          conv = (float*)(ws);                          // conv, then cm (in place)
    float*          rcm  = (float*)(ws + convBytes);
    unsigned short* Bt   = (unsigned short*)(ws + 2 * convBytes); // [208][2304] bf16
    unsigned short* embB = (unsigned short*)(ws + 2 * convBytes + btBytes);

    const bool pre = ws_size >= (2 * convBytes + btBytes + embBytes);

    // 0) optional bulk bf16 conversion of emb
    if (pre) {
        const int nvec = (BDIM * SDIM * HDIM) / 8;
        convert_emb_kernel<<<(nvec + 255) / 256, 256, 0, stream>>>(emb, embB);
    }
    // 1) weight transpose/pack to bf16
    {
        const int n = FPAD * KDIM;
        build_bt_kernel<<<(n + 255) / 256, 256, 0, stream>>>(conv_w, Bt);
    }
    // 2) conv GEMM via WMMA: 2048 M-tiles, one wave each
    if (pre)
        conv_wmma_kernel<true ><<<(BDIM * SDIM) / 16, 32, 0, stream>>>(emb, embB, Bt, conv_b, conv);
    else
        conv_wmma_kernel<false><<<(BDIM * SDIM) / 16, 32, 0, stream>>>(emb, embB, Bt, conv_b, conv);
    // 3) cummax scans
    cummax_kernel<<<BDIM, 256, 0, stream>>>(conv, rcm);
    // 4) gather + final linear (async-to-LDS path)
    logits_kernel<<<BDIM * SDIM, 128, 0, stream>>>(conv, rcm, L_w, L_b, out);
}
